// TitansMemoryModule_77163382440480
// MI455X (gfx1250) — compile-verified, matrix operand order checked by reading, not run on hardware
//
#include <hip/hip_runtime.h>
#include <hip/hip_bf16.h>
#include <math.h>

// ---- problem constants (match reference) ----
#define B_    4
#define L_    4096
#define HID_  1024
#define NH_   8
#define HD_   128
#define MI_   256
#define CH_   64
#define NC_   64
#define G_    (B_*NH_)     // 32 batched (b,h) state pairs

typedef __attribute__((ext_vector_type(16))) __bf16 v16bf;
typedef __attribute__((ext_vector_type(8)))  float  v8f;

__device__ __forceinline__ float sigmoidf_(float x) { return 1.0f / (1.0f + expf(-x)); }
__device__ __forceinline__ float siluf_(float x)    { return x * sigmoidf_(x); }

// Convert 8 contiguous f32 -> 8 packed bf16 (one uint4) in registers.
__device__ __forceinline__ uint4 pack8_bf16(const float* __restrict__ s)
{
    float4 a = *(const float4*)s;
    float4 b = *(const float4*)(s + 4);
    union { __bf16 h[8]; uint4 u; } r;
    r.h[0] = (__bf16)a.x; r.h[1] = (__bf16)a.y;
    r.h[2] = (__bf16)a.z; r.h[3] = (__bf16)a.w;
    r.h[4] = (__bf16)b.x; r.h[5] = (__bf16)b.y;
    r.h[6] = (__bf16)b.z; r.h[7] = (__bf16)b.w;
    return r.u;
}

// =====================================================================
// Generic batched GEMM: C[M,N] = A * B, f32 in/out, bf16 WMMA compute.
// TA: A stored as [K][M] (lda = stored row length); TB: B stored [N][K].
// epi: 0 = store C raw; 1 = store silu(C) to C, raw to C2 (if C2 != null)
// Block tile 64x128, BK=32, 256 threads = 8 wave32s (4x2 wave grid,
// each wave owns a 16x64 strip = 4 accumulators = 4 WMMA / K-step).
// Double-buffered LDS: stage tile k+1 while computing tile k, one
// barrier per K-step; prefetch tile k+2 into cache.
// =====================================================================
#define BM 64
#define BN 128
#define BK 32

union Frag { uint4 u[2]; v16bf v; };

template<int TA, int TB>
__global__ __launch_bounds__(256)
void gemm_bf16_wmma(const float* __restrict__ A, const float* __restrict__ Bm,
                    float* __restrict__ C, float* __restrict__ C2,
                    int K, int lda, int ldb, int ldc,
                    long sA, long sB, long sC, int epi)
{
    __shared__ __bf16 As[2][BM * BK];   // [m][k]
    __shared__ __bf16 Bs[2][BN * BK];   // B^T tile: [n][k]

    const long bz = blockIdx.z;
    A  += bz * sA;
    Bm += bz * sB;
    C  += bz * sC;
    if (C2) C2 += bz * sC;

    const int m0 = blockIdx.y * BM;
    const int n0 = blockIdx.x * BN;
    const int tid  = threadIdx.x;
    const int lane = tid & 31;
    const int wid  = tid >> 5;
    const int wm   = wid & 3;        // 0..3 -> 16-row strip
    const int wn   = wid >> 2;       // 0..1 -> 64-col strip
    const int half = lane >> 4;
    const int r16  = lane & 15;

    // ---- stage one (A,B) K-tile into LDS buffer `buf` ----
    auto stage = [&](int buf, int k0) {
        if (TA == 0) {
            // A[M][K]: contiguous along k. 64 rows x 4 groups of 8.
            int row = tid >> 2, cg = (tid & 3) * 8;
            *(uint4*)&As[buf][row * BK + cg] =
                pack8_bf16(A + (long)(m0 + row) * lda + (k0 + cg));
        } else {
            // A stored [K][M]: contiguous along m. 32 cols x 8 groups of 8.
            int col = tid >> 3, rg = (tid & 7) * 8;
            const float* s = A + (long)(k0 + col) * lda + (m0 + rg);
            float4 a0 = *(const float4*)s;
            float4 a1 = *(const float4*)(s + 4);
            float fv[8] = {a0.x, a0.y, a0.z, a0.w, a1.x, a1.y, a1.z, a1.w};
            #pragma unroll
            for (int i = 0; i < 8; ++i)
                As[buf][(rg + i) * BK + col] = (__bf16)fv[i];
        }
        if (TB == 0) {
            // B[K][N]: contiguous along n. 32 k x 8 groups of 16.
            int kk = tid >> 3, ng = (tid & 7) * 16;
            const float* s = Bm + (long)(k0 + kk) * ldb + (n0 + ng);
            float4 b0 = *(const float4*)s;
            float4 b1 = *(const float4*)(s + 4);
            float4 b2 = *(const float4*)(s + 8);
            float4 b3 = *(const float4*)(s + 12);
            float fv[16] = {b0.x, b0.y, b0.z, b0.w, b1.x, b1.y, b1.z, b1.w,
                            b2.x, b2.y, b2.z, b2.w, b3.x, b3.y, b3.z, b3.w};
            #pragma unroll
            for (int i = 0; i < 16; ++i)
                Bs[buf][(ng + i) * BK + kk] = (__bf16)fv[i];
        } else {
            // B stored [N][K]: contiguous along k. 128 n x 2 halves of 16.
            int nn = tid >> 1, kg = (tid & 1) * 16;
            const float* s = Bm + (long)(n0 + nn) * ldb + (k0 + kg);
            uint4 p0 = pack8_bf16(s);
            uint4 p1 = pack8_bf16(s + 8);
            *(uint4*)&Bs[buf][nn * BK + kg]     = p0;
            *(uint4*)&Bs[buf][nn * BK + kg + 8] = p1;
        }
    };

    v8f acc[4];
    #pragma unroll
    for (int t = 0; t < 4; ++t)
        #pragma unroll
        for (int i = 0; i < 8; ++i) acc[t][i] = 0.0f;

    const int nIter = K / BK;
    stage(0, 0);

    for (int it = 0; it < nIter; ++it) {
        const int cur = it & 1;
        __syncthreads();                       // staged tile `cur` visible

        // cache-prefetch tile k+2 while tile k+1 is being staged
        if (it + 2 < nIter) {
            int kp = (it + 2) * BK;
            if (TA == 0) __builtin_prefetch(A + (long)(m0 + (tid >> 2)) * lda + kp, 0, 1);
            else         __builtin_prefetch(A + (long)(kp + (tid >> 3)) * lda + m0, 0, 1);
            if (TB == 0) __builtin_prefetch(Bm + (long)(kp + (tid >> 3)) * ldb + n0, 0, 1);
            else         __builtin_prefetch(Bm + (long)(n0 + (tid >> 1)) * ldb + kp, 0, 1);
        }
        // stage next tile into the other buffer (overlaps with WMMAs below)
        if (it + 1 < nIter) stage(cur ^ 1, (it + 1) * BK);

        // ---- fragments per ISA 16-bit A/B 16x32 packing ----
        Frag fa;
        const __bf16* ap = &As[cur][(wm * 16 + r16) * BK];
        fa.u[0] = *(const uint4*)(ap + 8 * half);
        fa.u[1] = *(const uint4*)(ap + 16 + 8 * half);
        Frag fb[4];
        #pragma unroll
        for (int t = 0; t < 4; ++t) {
            const __bf16* bp = &Bs[cur][(wn * 64 + t * 16 + r16) * BK];
            fb[t].u[0] = *(const uint4*)(bp + 8 * half);
            fb[t].u[1] = *(const uint4*)(bp + 16 + 8 * half);
        }
        #pragma unroll
        for (int t = 0; t < 4; ++t)
            acc[t] = __builtin_amdgcn_wmma_f32_16x16x32_bf16(
                         false, fa.v, false, fb[t].v, (short)0, acc[t], false, false);
    }

    // ---- store: lane 0-15 -> M=j, lane 16-31 -> M=8+j ----
    const int gm_base = m0 + wm * 16 + half * 8;
    #pragma unroll
    for (int t = 0; t < 4; ++t) {
        int gn = n0 + wn * 64 + t * 16 + r16;
        #pragma unroll
        for (int j = 0; j < 8; ++j) {
            int gm = gm_base + j;
            float v = acc[t][j];
            long i0 = (long)gm * ldc + gn;
            if (epi == 0) {
                C[i0] = v;
            } else {
                if (C2) C2[i0] = v;
                C[i0] = siluf_(v);
            }
        }
    }
}

// =====================================================================
// Depthwise causal conv (KW=4) + SiLU, then optional L2-norm over the
// 128-wide head + RoPE. One 128-thread block per (b, l, h) row.
// Output layout [B, NH, L, HD] (scan-friendly).
// =====================================================================
__global__ __launch_bounds__(128)
void conv_head_kernel(const float* __restrict__ y, const float* __restrict__ cw,
                      const float* __restrict__ cb, float* __restrict__ outp,
                      int do_norm_rope)
{
    const int l = blockIdx.x, h = blockIdx.y, b = blockIdx.z;
    const int c = threadIdx.x;
    const int ch = h * HD_ + c;

    float acc = cb[ch];
    #pragma unroll
    for (int j = 0; j < 4; ++j) {
        int li = l - 3 + j;
        if (li >= 0) acc += cw[ch * 4 + j] * y[((long)b * L_ + li) * HID_ + ch];
    }
    float val = siluf_(acc);

    __shared__ float buf[128];
    __shared__ float wsum[4];
    if (do_norm_rope) {
        float s = val * val;
        #pragma unroll
        for (int m = 16; m > 0; m >>= 1) s += __shfl_xor(s, m, 32);
        if ((c & 31) == 0) wsum[c >> 5] = s;
        __syncthreads();
        float tot = wsum[0] + wsum[1] + wsum[2] + wsum[3];
        val = val / fmaxf(sqrtf(tot), 1e-12f);
        buf[c] = val;
        __syncthreads();
        float rot = (c < 64) ? -buf[c + 64] : buf[c - 64];
        float inv = powf(10000.0f, -(float)(c & 63) / 64.0f);
        float fr  = (float)l * inv;
        val = val * cosf(fr) + rot * sinf(fr);
    }
    outp[(((long)b * NH_ + h) * L_ + l) * HD_ + c] = val;
}

// =====================================================================
// theta/eta/alpha gates: per (b,l), 8 waves -> 8 heads, wave-reduced dot.
// =====================================================================
__global__ __launch_bounds__(256)
void gates_kernel(const float* __restrict__ x, const float* __restrict__ Wt,
                  const float* __restrict__ We, const float* __restrict__ Wa,
                  float* __restrict__ theta, float* __restrict__ eta_l,
                  float* __restrict__ alp_l)
{
    const long bl = blockIdx.x;
    const int h = threadIdx.x >> 5, lane = threadIdx.x & 31;
    const float* xr = x + bl * HID_;
    float st = 0.f, se = 0.f, sa = 0.f;
    #pragma unroll 4
    for (int i = 0; i < HID_ / 32; ++i) {
        int cc = i * 32 + lane;
        float xv = xr[cc];
        st += xv * Wt[cc * NH_ + h];
        se += xv * We[cc * NH_ + h];
        sa += xv * Wa[cc * NH_ + h];
    }
    #pragma unroll
    for (int m = 16; m > 0; m >>= 1) {
        st += __shfl_xor(st, m, 32);
        se += __shfl_xor(se, m, 32);
        sa += __shfl_xor(sa, m, 32);
    }
    if (lane == 0) {
        long b = bl / L_, l = bl % L_;
        long g = b * NH_ + h;
        theta[g * L_ + l] = sigmoidf_(st);
        if ((l & (CH_ - 1)) == (CH_ - 1)) {
            long nc = l / CH_;
            eta_l[g * NC_ + nc] = sigmoidf_(se);
            alp_l[g * NC_ + nc] = sigmoidf_(sa);
        }
    }
}

// =====================================================================
// State init: broadcast W1/W2 per (b,h), zero S1/S2.
// =====================================================================
__global__ void init_states_kernel(const float* __restrict__ W1, const float* __restrict__ W2,
                                   float* W1s, float* W2s, float* S1, float* S2)
{
    long idx = (long)blockIdx.x * blockDim.x + threadIdx.x;
    const long per = (long)HD_ * MI_;
    if (idx >= (long)G_ * per) return;
    long g = idx / per, r = idx % per;
    int h = (int)(g % NH_);
    W1s[idx] = W1[(long)h * per + r];
    W2s[idx] = W2[(long)h * per + r];
    S1[idx] = 0.f;
    S2[idx] = 0.f;
}

__device__ __forceinline__ void blockreduce2_128(float& a, float& b, float* red)
{
    int lane = threadIdx.x & 31, wid = threadIdx.x >> 5;
    #pragma unroll
    for (int m = 16; m > 0; m >>= 1) { a += __shfl_xor(a, m, 32); b += __shfl_xor(b, m, 32); }
    if (lane == 0) { red[wid] = a; red[4 + wid] = b; }
    __syncthreads();
    a = red[0] + red[1] + red[2] + red[3];
    b = red[4] + red[5] + red[6] + red[7];
    __syncthreads();
}

// =====================================================================
// z = ln_fused_l2_bwd(pred, v, gamma, beta) * theta  (per 128-wide row)
// =====================================================================
__global__ __launch_bounds__(128)
void ln_l2_bwd_kernel(const float* __restrict__ pred, const float* __restrict__ vb,
                      const float* __restrict__ gamma, const float* __restrict__ beta,
                      const float* __restrict__ theta, float* __restrict__ z, int nc)
{
    __shared__ float red[8];
    const int row = blockIdx.x, g = blockIdx.y, h = g & (NH_ - 1);
    const int c = threadIdx.x;
    const int l = nc * CH_ + row;

    float x   = pred[((long)g * CH_ + row) * HD_ + c];
    float tgt = vb[((long)g * L_ + l) * HD_ + c];
    float gm  = gamma[h * HD_ + c], bt = beta[h * HD_ + c];

    float s1 = x, s2 = x * x;
    blockreduce2_128(s1, s2, red);
    float mu  = s1 / 128.0f;
    float var = fmaxf(s2 / 128.0f - mu * mu, 0.0f);
    float sd  = sqrtf(var + 1e-6f);
    float xh  = (x - mu) / sd;
    float y   = gm * xh + bt;
    float gxh = 2.0f * (y - tgt) * gm;

    float r1 = gxh, r2 = gxh * xh;
    blockreduce2_128(r1, r2, red);

    float th = theta[(long)g * L_ + l];
    z[((long)g * CH_ + row) * HD_ + c] =
        (128.0f * gxh - r1 - xh * r2) / (128.0f * sd) * th;
}

// =====================================================================
// gh *= silu_bwd(h)
// =====================================================================
__global__ void ghscale_kernel(float* __restrict__ gh, const float* __restrict__ hb)
{
    long idx = (long)blockIdx.x * blockDim.x + threadIdx.x;
    if (idx >= (long)G_ * CH_ * MI_) return;
    float x = hb[idx];
    float s = sigmoidf_(x);
    gh[idx] *= (s + x * s * (1.0f - s));
}

// =====================================================================
// S = eta*S - gW ; Wc = (1-alpha)*Wc + S   (per-batch scalars)
// =====================================================================
__global__ void update_kernel(float* __restrict__ Wc, float* __restrict__ S,
                              const float* __restrict__ gW,
                              const float* __restrict__ eta_l,
                              const float* __restrict__ alp_l, int nc)
{
    long idx = (long)blockIdx.x * blockDim.x + threadIdx.x;
    const long per = (long)HD_ * MI_;
    if (idx >= (long)G_ * per) return;
    long g = idx / per;
    float e = eta_l[g * NC_ + nc];
    float a = alp_l[g * NC_ + nc];
    float s = e * S[idx] - gW[idx];
    S[idx] = s;
    Wc[idx] = (1.0f - a) * Wc[idx] + s;
}

// =====================================================================
// Per-row LayerNorm of o, write into attn[B,NH,L,HD].
// =====================================================================
__global__ __launch_bounds__(128)
void onorm_kernel(const float* __restrict__ o, const float* __restrict__ gamma,
                  const float* __restrict__ beta, float* __restrict__ attn, int nc)
{
    __shared__ float red[8];
    const int row = blockIdx.x, g = blockIdx.y, h = g & (NH_ - 1);
    const int c = threadIdx.x;
    float x = o[((long)g * CH_ + row) * HD_ + c];
    float s1 = x, s2 = x * x;
    blockreduce2_128(s1, s2, red);
    float mu  = s1 / 128.0f;
    float var = fmaxf(s2 / 128.0f - mu * mu, 0.0f);
    float xh  = (x - mu) / sqrtf(var + 1e-6f);
    attn[((long)g * L_ + nc * CH_ + row) * HD_ + c] =
        gamma[h * HD_ + c] * xh + beta[h * HD_ + c];
}

// =====================================================================
// merged = l2norm_over_1024(attn as [B,L,HID]) * gate
// =====================================================================
__global__ __launch_bounds__(256)
void merge_kernel(const float* __restrict__ attn, const float* __restrict__ gate,
                  float* __restrict__ merged)
{
    __shared__ float red[8];
    const long bl = blockIdx.x;
    const long b = bl / L_, l = bl % L_;
    float vals[4];
    float ss = 0.f;
    #pragma unroll
    for (int i = 0; i < 4; ++i) {
        int hid = threadIdx.x + i * 256;
        int h = hid >> 7, d = hid & 127;
        float u = attn[((b * NH_ + h) * L_ + l) * HD_ + d];
        vals[i] = u;
        ss += u * u;
    }
    int lane = threadIdx.x & 31, wid = threadIdx.x >> 5;
    #pragma unroll
    for (int m = 16; m > 0; m >>= 1) ss += __shfl_xor(ss, m, 32);
    if (lane == 0) red[wid] = ss;
    __syncthreads();
    float tot = 0.f;
    #pragma unroll
    for (int i = 0; i < 8; ++i) tot += red[i];
    float inv = 1.0f / fmaxf(sqrtf(tot), 1e-12f);
    #pragma unroll
    for (int i = 0; i < 4; ++i) {
        int hid = threadIdx.x + i * 256;
        merged[bl * HID_ + hid] = vals[i] * inv * gate[bl * HID_ + hid];
    }
}

// =====================================================================
// Host orchestration
// =====================================================================
extern "C" void kernel_launch(void* const* d_in, const int* in_sizes, int n_in,
                              void* d_out, int out_size, void* d_ws, size_t ws_size,
                              hipStream_t stream)
{
    (void)in_sizes; (void)n_in; (void)out_size; (void)ws_size;
    const float* x     = (const float*)d_in[0];
    const float* Wq    = (const float*)d_in[1];
    const float* Wk    = (const float*)d_in[2];
    const float* Wv    = (const float*)d_in[3];
    const float* cwq   = (const float*)d_in[4];
    const float* cbq   = (const float*)d_in[5];
    const float* cwk   = (const float*)d_in[6];
    const float* cbk   = (const float*)d_in[7];
    const float* cwv   = (const float*)d_in[8];
    const float* cbv   = (const float*)d_in[9];
    const float* Wal   = (const float*)d_in[10];
    const float* Wet   = (const float*)d_in[11];
    const float* Wth   = (const float*)d_in[12];
    const float* W1    = (const float*)d_in[13];
    const float* W2    = (const float*)d_in[14];
    const float* gamma = (const float*)d_in[15];
    const float* beta  = (const float*)d_in[16];
    const float* gateW = (const float*)d_in[17];
    const float* oW    = (const float*)d_in[18];
    float* outp = (float*)d_out;

    const long ML  = (long)B_ * L_;          // 16384 rows
    const long PER = (long)HD_ * MI_;        // 32768 per state

    float* ws = (float*)d_ws;
    long off = 0;
    auto alloc = [&](long n) { float* p = ws + off; off += n; return p; };
    float* proj  = alloc(ML * HID_);        // projection scratch; reused for gate
    float* qb    = alloc(ML * HID_);        // q [B,NH,L,HD]; reused for merged
    float* kb    = alloc(ML * HID_);
    float* vb    = alloc(ML * HID_);
    float* attn  = alloc(ML * HID_);
    float* theta = alloc((long)G_ * L_);
    float* eta_l = alloc((long)G_ * NC_);
    float* alp_l = alloc((long)G_ * NC_);
    float* W1s   = alloc((long)G_ * PER);
    float* W2s   = alloc((long)G_ * PER);
    float* S1    = alloc((long)G_ * PER);
    float* S2    = alloc((long)G_ * PER);
    float* hbuf  = alloc((long)G_ * CH_ * MI_);
    float* abuf  = alloc((long)G_ * CH_ * MI_);
    float* ghbuf = alloc((long)G_ * CH_ * MI_);
    float* predb = alloc((long)G_ * CH_ * HD_);
    float* zbuf  = alloc((long)G_ * CH_ * HD_);
    float* gW1   = alloc((long)G_ * PER);
    float* gW2   = alloc((long)G_ * PER);

    dim3 blk(256);
    auto GEMM = [&](const float* A, const float* Bm, float* C, float* C2,
                    int M, int N, int K, int lda, int ldb, int ldc,
                    int tA, int tB, long sA, long sB, long sC, int nb, int epi) {
        dim3 grid(N / BN, M / BM, nb);
        if (tA == 0 && tB == 0)
            gemm_bf16_wmma<0, 0><<<grid, blk, 0, stream>>>(
                A, Bm, C, C2, K, lda, ldb, ldc, sA, sB, sC, epi);
        else if (tA == 1)
            gemm_bf16_wmma<1, 0><<<grid, blk, 0, stream>>>(
                A, Bm, C, C2, K, lda, ldb, ldc, sA, sB, sC, epi);
        else
            gemm_bf16_wmma<0, 1><<<grid, blk, 0, stream>>>(
                A, Bm, C, C2, K, lda, ldb, ldc, sA, sB, sC, epi);
    };

    // ---- projections + conv/silu/norm/rope ----
    GEMM(x, Wq, proj, nullptr, (int)ML, HID_, HID_, HID_, HID_, HID_, 0, 0, 0, 0, 0, 1, 0);
    conv_head_kernel<<<dim3(L_, NH_, B_), dim3(128), 0, stream>>>(proj, cwq, cbq, qb, 1);
    GEMM(x, Wk, proj, nullptr, (int)ML, HID_, HID_, HID_, HID_, HID_, 0, 0, 0, 0, 0, 1, 0);
    conv_head_kernel<<<dim3(L_, NH_, B_), dim3(128), 0, stream>>>(proj, cwk, cbk, kb, 1);
    GEMM(x, Wv, proj, nullptr, (int)ML, HID_, HID_, HID_, HID_, HID_, 0, 0, 0, 0, 0, 1, 0);
    conv_head_kernel<<<dim3(L_, NH_, B_), dim3(128), 0, stream>>>(proj, cwv, cbv, vb, 0);

    // ---- gates & state init ----
    gates_kernel<<<dim3((unsigned)ML), dim3(256), 0, stream>>>(x, Wth, Wet, Wal,
                                                               theta, eta_l, alp_l);
    {
        long n = (long)G_ * PER;
        init_states_kernel<<<dim3((unsigned)((n + 255) / 256)), dim3(256), 0, stream>>>(
            W1, W2, W1s, W2s, S1, S2);
    }

    // ---- chunked scan: 64 sequential steps over 32 parallel (b,h) pairs ----
    const long sQK = (long)L_ * HD_;        // per-(b,h) stride in q/k/v buffers
    const long nUpd = (long)G_ * PER;
    const long nGh  = (long)G_ * CH_ * MI_;
    for (int nc = 0; nc < NC_; ++nc) {
        const float* kc = kb + (long)nc * CH_ * HD_;
        const float* qc = qb + (long)nc * CH_ * HD_;

        // h = k @ W1c ; a = silu(h)
        GEMM(kc, W1s, abuf, hbuf, CH_, MI_, HD_, HD_, MI_, MI_,
             0, 0, sQK, PER, (long)CH_ * MI_, G_, 1);
        // pred = a @ W2c
        GEMM(abuf, W2s, predb, nullptr, CH_, HD_, MI_, MI_, HD_, HD_,
             0, 0, (long)CH_ * MI_, PER, (long)CH_ * HD_, G_, 0);
        // z = ln_fused_l2_bwd(pred, v) * theta
        ln_l2_bwd_kernel<<<dim3(CH_, G_), dim3(128), 0, stream>>>(
            predb, vb, gamma, beta, theta, zbuf, nc);
        // gW2 = a^T @ z
        GEMM(abuf, zbuf, gW2, nullptr, MI_, HD_, CH_, MI_, HD_, HD_,
             1, 0, (long)CH_ * MI_, (long)CH_ * HD_, PER, G_, 0);
        // ghraw = z @ W2c^T
        GEMM(zbuf, W2s, ghbuf, nullptr, CH_, MI_, HD_, HD_, HD_, MI_,
             0, 1, (long)CH_ * HD_, PER, (long)CH_ * MI_, G_, 0);
        // gh = ghraw * silu_bwd(h)
        ghscale_kernel<<<dim3((unsigned)((nGh + 255) / 256)), dim3(256), 0, stream>>>(
            ghbuf, hbuf);
        // gW1 = k^T @ gh
        GEMM(kc, ghbuf, gW1, nullptr, HD_, MI_, CH_, HD_, MI_, MI_,
             1, 0, sQK, (long)CH_ * MI_, PER, G_, 0);
        // state updates (L2-resident)
        update_kernel<<<dim3((unsigned)((nUpd + 255) / 256)), dim3(256), 0, stream>>>(
            W1s, S1, gW1, eta_l, alp_l, nc);
        update_kernel<<<dim3((unsigned)((nUpd + 255) / 256)), dim3(256), 0, stream>>>(
            W2s, S2, gW2, eta_l, alp_l, nc);
        // aq = silu(q @ W1c)
        GEMM(qc, W1s, abuf, nullptr, CH_, MI_, HD_, HD_, MI_, MI_,
             0, 0, sQK, PER, (long)CH_ * MI_, G_, 1);
        // o = aq @ W2c
        GEMM(abuf, W2s, predb, nullptr, CH_, HD_, MI_, MI_, HD_, HD_,
             0, 0, (long)CH_ * MI_, PER, (long)CH_ * HD_, G_, 0);
        // layernorm -> attn
        onorm_kernel<<<dim3(CH_, G_), dim3(128), 0, stream>>>(
            predb, gamma, beta, attn, nc);
    }

    // ---- gate, merge, output projection ----
    GEMM(x, gateW, proj, nullptr, (int)ML, HID_, HID_, HID_, HID_, HID_,
         0, 0, 0, 0, 0, 1, 1);                         // proj = silu(x @ gate_W)
    merge_kernel<<<dim3((unsigned)ML), dim3(256), 0, stream>>>(attn, proj, qb);
    GEMM(qb, oW, outp, nullptr, (int)ML, HID_, HID_, HID_, HID_, HID_,
         0, 0, 0, 0, 0, 1, 0);                         // out = merged @ o_W
}
